// DS3L_DGCNN_cls_46840913330492
// MI455X (gfx1250) — matrix operand order repaired and checked by reading, not compiled, not run on hardware
//
#include <hip/hip_runtime.h>
#include <hip/hip_bf16.h>

#define NPTS  1024
#define KNN   20
#define NKTOT (NPTS*KNN)
#define BATCH 8

typedef _Float16 v16h __attribute__((ext_vector_type(16)));
typedef float    v8f  __attribute__((ext_vector_type(8)));
typedef float    v2f  __attribute__((ext_vector_type(2)));

union FragH { v16h h; unsigned u[8]; };

__device__ __forceinline__ float lrelu_f(float v){ return v >= 0.f ? v : 0.2f*v; }

// ---------------------------------------------------------------------------
// 1) per-point squared norms  xx[b][n] = sum_c x[b][c][n]^2
// ---------------------------------------------------------------------------
template<int CIN>
__global__ void xx_kernel(const float* __restrict__ x, float* __restrict__ xx){
  int t = blockIdx.x*blockDim.x + threadIdx.x;
  if (t >= BATCH*NPTS) return;
  int b = t / NPTS, n = t % NPTS;
  const float* xb = x + (size_t)b*CIN*NPTS + n;
  float s = 0.f;
  #pragma unroll 4
  for (int c = 0; c < CIN; ++c){ float v = xb[(size_t)c*NPTS]; s += v*v; }
  xx[t] = s;
}

// ---------------------------------------------------------------------------
// 2) pairwise matrix via f32 WMMA: pd[n][m] = 2*dot(x_n,x_m) - xx_n - xx_m
//    Wave owns 1 n-tile x 4 m-tiles: A fragment reused across 4 WMMAs.
// ---------------------------------------------------------------------------
template<int CIN>
__global__ __launch_bounds__(256)
void pd_kernel(const float* __restrict__ x, const float* __restrict__ xx,
               float* __restrict__ pd){
  const int b    = blockIdx.z;
  const int wave = threadIdx.x >> 5;
  const int lane = threadIdx.x & 31;
  const int hi   = lane >> 4;
  const int lm   = lane & 15;
  const int grp  = blockIdx.x*8 + wave;      // 16 groups of 4 m-tiles
  const int m0   = grp*64;
  const int n0   = blockIdx.y*16;
  const float* xb = x + (size_t)b*CIN*NPTS;

  v8f acc0 = {}, acc1 = {}, acc2 = {}, acc3 = {};
  constexpr int NCH = (CIN + 3) >> 2;
  #pragma unroll
  for (int ch = 0; ch < NCH; ++ch){
    const int c0  = ch*4;
    const int ka0 = c0 + (hi ? 2 : 0);
    const int ka1 = ka0 + 1;
    v2f a;
    a.x = (ka0 < CIN) ? xb[(size_t)ka0*NPTS + n0 + lm] : 0.f;
    a.y = (ka1 < CIN) ? xb[(size_t)ka1*NPTS + n0 + lm] : 0.f;
    v2f b0, b1, b2, b3;
    b0.x = (ka0 < CIN) ? xb[(size_t)ka0*NPTS + m0 +  0 + lm] : 0.f;
    b0.y = (ka1 < CIN) ? xb[(size_t)ka1*NPTS + m0 +  0 + lm] : 0.f;
    b1.x = (ka0 < CIN) ? xb[(size_t)ka0*NPTS + m0 + 16 + lm] : 0.f;
    b1.y = (ka1 < CIN) ? xb[(size_t)ka1*NPTS + m0 + 16 + lm] : 0.f;
    b2.x = (ka0 < CIN) ? xb[(size_t)ka0*NPTS + m0 + 32 + lm] : 0.f;
    b2.y = (ka1 < CIN) ? xb[(size_t)ka1*NPTS + m0 + 32 + lm] : 0.f;
    b3.x = (ka0 < CIN) ? xb[(size_t)ka0*NPTS + m0 + 48 + lm] : 0.f;
    b3.y = (ka1 < CIN) ? xb[(size_t)ka1*NPTS + m0 + 48 + lm] : 0.f;
    acc0 = __builtin_amdgcn_wmma_f32_16x16x4_f32(false, a, false, b0, (short)0, acc0, false, false);
    acc1 = __builtin_amdgcn_wmma_f32_16x16x4_f32(false, a, false, b1, (short)0, acc1, false, false);
    acc2 = __builtin_amdgcn_wmma_f32_16x16x4_f32(false, a, false, b2, (short)0, acc2, false, false);
    acc3 = __builtin_amdgcn_wmma_f32_16x16x4_f32(false, a, false, b3, (short)0, acc3, false, false);
  }
  const float xxm0 = xx[b*NPTS + m0 +  0 + lm];
  const float xxm1 = xx[b*NPTS + m0 + 16 + lm];
  const float xxm2 = xx[b*NPTS + m0 + 32 + lm];
  const float xxm3 = xx[b*NPTS + m0 + 48 + lm];
  #pragma unroll
  for (int v = 0; v < 8; ++v){
    int n = n0 + v + hi*8;
    float xxn = xx[b*NPTS + n];
    float* row = pd + ((size_t)b*NPTS + n)*NPTS + m0 + lm;
    row[ 0] = 2.f*acc0[v] - xxn - xxm0;
    row[16] = 2.f*acc1[v] - xxn - xxm1;
    row[32] = 2.f*acc2[v] - xxn - xxm2;
    row[48] = 2.f*acc3[v] - xxn - xxm3;
  }
}

// ---------------------------------------------------------------------------
// 3) top-K(=20) indices per row (insertion select; ties keep lower index)
// ---------------------------------------------------------------------------
__global__ void topk_kernel(const float* __restrict__ pd, int* __restrict__ idx){
  int t = blockIdx.x*blockDim.x + threadIdx.x;
  if (t >= BATCH*NPTS) return;
  const float* row = pd + (size_t)t*NPTS;
  float bv[KNN]; int bi[KNN];
  #pragma unroll
  for (int i = 0; i < KNN; ++i){ bv[i] = -3.4e38f; bi[i] = 0; }
  for (int m = 0; m < NPTS; ++m){
    float v = row[m];
    if (v > bv[KNN-1]){
      int p = KNN-1;
      while (p > 0 && bv[p-1] < v){ bv[p] = bv[p-1]; bi[p] = bi[p-1]; --p; }
      bv[p] = v; bi[p] = m;
    }
  }
  #pragma unroll
  for (int i = 0; i < KNN; ++i) idx[(size_t)t*KNN + i] = bi[i];
}

// ---------------------------------------------------------------------------
__global__ void zero_kernel(float* __restrict__ p, int n){
  int t = blockIdx.x*blockDim.x + threadIdx.x;
  if (t < n) p[t] = 0.f;
}

// ---------------------------------------------------------------------------
// 4) fused gather + GEMM:  y[b][o][nk] = sum_c W[o][c] * F[c][nk]
//    F[c][nk]: c<CIN -> x[c][idx]-x[c][n] ; else -> x[c-CIN][n]
//    f16 in / f32 accum. Block = 8 waves covering 64 O x 128 NK; each wave
//    owns a 2x2 grid of 16x16 accumulator tiles (4 WMMAs per K-chunk).
// ---------------------------------------------------------------------------
template<int CIN, int COUT>
__global__ __launch_bounds__(256)
void edgeconv_gemm(const float* __restrict__ x, const int* __restrict__ idx,
                   const float* __restrict__ w, float* __restrict__ y,
                   float* __restrict__ sum, float* __restrict__ sumsq){
  const int b    = blockIdx.z;
  const int nkB  = blockIdx.x * 128;
  const int wave = threadIdx.x >> 5;
  const int lane = threadIdx.x & 31;
  const int hi   = lane >> 4;
  const int lm   = lane & 15;
  const int wo   = wave >> 2;                // 0..1  -> 32 O rows each
  const int wn   = wave & 3;                 // 0..3  -> 32 NK cols each
  const int oB   = blockIdx.y*64;

  __shared__ _Float16 sF[128][34];           // [nk col][k]
  __shared__ _Float16 sW[64][34];            // [o row][k]

  constexpr int twoC   = 2*CIN;
  constexpr int NCHUNK = (twoC + 31) >> 5;
  const float* xb  = x   + (size_t)b*CIN*NPTS;
  const int*  idxb = idx + (size_t)b*NPTS*KNN;

  v8f a00 = {}, a01 = {}, a10 = {}, a11 = {};   // [o-tile][nk-tile]
  for (int ch = 0; ch < NCHUNK; ++ch){
    const int k0 = ch*32;
    // stage feature tile: 128 cols x 32 k
    for (int t = threadIdx.x; t < 128*32; t += 256){
      int col = t >> 5, kk = t & 31, kr = k0 + kk;
      int nk = nkB + col, n = nk / KNN, kn = nk % KNN;
      float v = 0.f;
      if (kr < twoC){
        if (kr < CIN){
          int j = idxb[n*KNN + kn];
          v = xb[(size_t)kr*NPTS + j] - xb[(size_t)kr*NPTS + n];
        } else {
          v = xb[(size_t)(kr - CIN)*NPTS + n];
        }
      }
      sF[col][kk] = (_Float16)v;
    }
    // stage weight tile: 64 rows x 32 k
    for (int t = threadIdx.x; t < 64*32; t += 256){
      int m = t >> 5, kk = t & 31, kr = k0 + kk;
      float v = (kr < twoC) ? w[(size_t)(oB + m)*twoC + kr] : 0.f;
      sW[m][kk] = (_Float16)v;
    }
    __syncthreads();
    {
      FragH a0, a1, f0, f1;
      #pragma unroll
      for (int v = 0; v < 8; ++v){
        int kA = ((v & 3) << 1) + (v >= 4 ? 16 : 0) + (hi ? 8 : 0);
        a0.u[v] = *(const unsigned*)&sW[wo*32 +      lm][kA];
        a1.u[v] = *(const unsigned*)&sW[wo*32 + 16 + lm][kA];
        int kB = ((v & 3) << 1) + (v >= 4 ? 16 : 0) + (hi ? 8 : 0);
        f0.u[v] = *(const unsigned*)&sF[wn*32 +      lm][kB];
        f1.u[v] = *(const unsigned*)&sF[wn*32 + 16 + lm][kB];
      }
      a00 = __builtin_amdgcn_wmma_f32_16x16x32_f16(false, a0.h, false, f0.h, (short)0, a00, false, false);
      a01 = __builtin_amdgcn_wmma_f32_16x16x32_f16(false, a0.h, false, f1.h, (short)0, a01, false, false);
      a10 = __builtin_amdgcn_wmma_f32_16x16x32_f16(false, a1.h, false, f0.h, (short)0, a10, false, false);
      a11 = __builtin_amdgcn_wmma_f32_16x16x32_f16(false, a1.h, false, f1.h, (short)0, a11, false, false);
    }
    __syncthreads();
  }

  const int nk0 = nkB + wn*32;
  #pragma unroll
  for (int v = 0; v < 8; ++v){
    int o0 = oB + wo*32 + v + hi*8;          // first o-tile row
    int o1 = o0 + 16;                        // second o-tile row
    float v00 = a00[v], v01 = a01[v], v10 = a10[v], v11 = a11[v];
    float* y0 = y + ((size_t)b*COUT + o0)*NKTOT + nk0 + lm;
    float* y1 = y + ((size_t)b*COUT + o1)*NKTOT + nk0 + lm;
    y0[ 0] = v00; y0[16] = v01;
    y1[ 0] = v10; y1[16] = v11;
    // fold the two nk tiles, then tree-reduce across the 16 columns
    float s0 = v00 + v01, q0 = v00*v00 + v01*v01;
    float s1 = v10 + v11, q1 = v10*v10 + v11*v11;
    #pragma unroll
    for (int off = 8; off >= 1; off >>= 1){
      s0 += __shfl_xor(s0, off, 16);  q0 += __shfl_xor(q0, off, 16);
      s1 += __shfl_xor(s1, off, 16);  q1 += __shfl_xor(q1, off, 16);
    }
    if (lm == 0){
      atomicAdd(&sum[o0], s0); atomicAdd(&sumsq[o0], q0);
      atomicAdd(&sum[o1], s1); atomicAdd(&sumsq[o1], q1);
    }
  }
}

// ---------------------------------------------------------------------------
// 5) BN(train) + lrelu + max over k   -> xout[b][o][n]
// ---------------------------------------------------------------------------
__global__ void bn_lrelu_maxk(const float* __restrict__ y, const float* __restrict__ sum,
                              const float* __restrict__ sumsq, const float* __restrict__ bw,
                              const float* __restrict__ bb, float* __restrict__ out, int Cout){
  int t = blockIdx.x*blockDim.x + threadIdx.x;
  if (t >= BATCH*Cout*NPTS) return;
  int n = t % NPTS, bo = t / NPTS, o = bo % Cout, b = bo / Cout;
  const float cnt = (float)BATCH * (float)NKTOT;
  float m  = sum[o] / cnt;
  float vr = sumsq[o] / cnt - m*m;
  float sc = rsqrtf(vr + 1e-5f) * bw[o];
  float sh = bb[o] - m*sc;
  const float* yp = y + ((size_t)b*Cout + o)*NKTOT + (size_t)n*KNN;
  float mx = -3.4e38f;
  #pragma unroll
  for (int k = 0; k < KNN; ++k) mx = fmaxf(mx, lrelu_f(yp[k]*sc + sh));
  out[((size_t)b*Cout + o)*NPTS + n] = mx;
}

// ---------------------------------------------------------------------------
// 6) conv5 GEMM over implicit concat(x1..x4): y5[b][o][n], O=1024, C=512
//    Same 2x2-tile-per-wave scheme; block covers 64 O x 128 N.
// ---------------------------------------------------------------------------
__device__ __forceinline__ float cat_feat(const float* x1, const float* x2,
                                          const float* x3, const float* x4,
                                          int b, int c, int n){
  if (c < 64)  return x1[((size_t)b*64  +  c       )*NPTS + n];
  if (c < 128) return x2[((size_t)b*64  + (c - 64 ))*NPTS + n];
  if (c < 256) return x3[((size_t)b*128 + (c - 128))*NPTS + n];
  return              x4[((size_t)b*256 + (c - 256))*NPTS + n];
}

__global__ __launch_bounds__(256)
void conv5_gemm(const float* __restrict__ x1, const float* __restrict__ x2,
                const float* __restrict__ x3, const float* __restrict__ x4,
                const float* __restrict__ w, float* __restrict__ y5,
                float* __restrict__ sum, float* __restrict__ sumsq){
  const int b    = blockIdx.z;
  const int nB   = blockIdx.x * 128;
  const int wave = threadIdx.x >> 5;
  const int lane = threadIdx.x & 31;
  const int hi   = lane >> 4;
  const int lm   = lane & 15;
  const int wo   = wave >> 2;
  const int wn   = wave & 3;
  const int oB   = blockIdx.y*64;

  __shared__ _Float16 sF[128][34];
  __shared__ _Float16 sW[64][34];

  v8f a00 = {}, a01 = {}, a10 = {}, a11 = {};
  #pragma unroll 1
  for (int ch = 0; ch < 512/32; ++ch){
    const int k0 = ch*32;
    for (int t = threadIdx.x; t < 128*32; t += 256){
      int col = t >> 5, kk = t & 31;
      sF[col][kk] = (_Float16)cat_feat(x1,x2,x3,x4, b, k0+kk, nB+col);
    }
    for (int t = threadIdx.x; t < 64*32; t += 256){
      int m = t >> 5, kk = t & 31;
      sW[m][kk] = (_Float16)w[(size_t)(oB + m)*512 + k0 + kk];
    }
    __syncthreads();
    {
      FragH a0, a1, f0, f1;
      #pragma unroll
      for (int v = 0; v < 8; ++v){
        int kA = ((v & 3) << 1) + (v >= 4 ? 16 : 0) + (hi ? 8 : 0);
        a0.u[v] = *(const unsigned*)&sW[wo*32 +      lm][kA];
        a1.u[v] = *(const unsigned*)&sW[wo*32 + 16 + lm][kA];
        int kB = ((v & 3) << 1) + (v >= 4 ? 16 : 0) + (hi ? 8 : 0);
        f0.u[v] = *(const unsigned*)&sF[wn*32 +      lm][kB];
        f1.u[v] = *(const unsigned*)&sF[wn*32 + 16 + lm][kB];
      }
      a00 = __builtin_amdgcn_wmma_f32_16x16x32_f16(false, a0.h, false, f0.h, (short)0, a00, false, false);
      a01 = __builtin_amdgcn_wmma_f32_16x16x32_f16(false, a0.h, false, f1.h, (short)0, a01, false, false);
      a10 = __builtin_amdgcn_wmma_f32_16x16x32_f16(false, a1.h, false, f0.h, (short)0, a10, false, false);
      a11 = __builtin_amdgcn_wmma_f32_16x16x32_f16(false, a1.h, false, f1.h, (short)0, a11, false, false);
    }
    __syncthreads();
  }

  const int n0 = nB + wn*32;
  #pragma unroll
  for (int v = 0; v < 8; ++v){
    int o0 = oB + wo*32 + v + hi*8;
    int o1 = o0 + 16;
    float v00 = a00[v], v01 = a01[v], v10 = a10[v], v11 = a11[v];
    float* y0 = y5 + ((size_t)b*1024 + o0)*NPTS + n0 + lm;
    float* y1 = y5 + ((size_t)b*1024 + o1)*NPTS + n0 + lm;
    y0[ 0] = v00; y0[16] = v01;
    y1[ 0] = v10; y1[16] = v11;
    float s0 = v00 + v01, q0 = v00*v00 + v01*v01;
    float s1 = v10 + v11, q1 = v10*v10 + v11*v11;
    #pragma unroll
    for (int off = 8; off >= 1; off >>= 1){
      s0 += __shfl_xor(s0, off, 16);  q0 += __shfl_xor(q0, off, 16);
      s1 += __shfl_xor(s1, off, 16);  q1 += __shfl_xor(q1, off, 16);
    }
    if (lm == 0){
      atomicAdd(&sum[o0], s0); atomicAdd(&sumsq[o0], q0);
      atomicAdd(&sum[o1], s1); atomicAdd(&sumsq[o1], q1);
    }
  }
}

// ---------------------------------------------------------------------------
// 7) BN + lrelu + (max,mean) pool over n -> feat[b][0..1023]=max, [1024..]=mean
// ---------------------------------------------------------------------------
__global__ void bn_lrelu_pool(const float* __restrict__ y5, const float* __restrict__ sum,
                              const float* __restrict__ sumsq, const float* __restrict__ bw,
                              const float* __restrict__ bb, float* __restrict__ feat){
  int t = blockIdx.x*blockDim.x + threadIdx.x;
  if (t >= BATCH*1024) return;
  int o = t % 1024, b = t / 1024;
  const float cnt = (float)BATCH * (float)NPTS;
  float m  = sum[o] / cnt;
  float vr = sumsq[o] / cnt - m*m;
  float sc = rsqrtf(vr + 1e-5f) * bw[o];
  float sh = bb[o] - m*sc;
  const float* yp = y5 + ((size_t)b*1024 + o)*NPTS;
  float mx = -3.4e38f, sm = 0.f;
  for (int n = 0; n < NPTS; ++n){
    float v = lrelu_f(yp[n]*sc + sh);
    mx = fmaxf(mx, v); sm += v;
  }
  feat[(size_t)b*2048 + o]        = mx;
  feat[(size_t)b*2048 + 1024 + o] = sm * (1.f/NPTS);
}

// ---------------------------------------------------------------------------
// 8) FC layers (batch 8 — scalar) + batch-BN + lrelu
// ---------------------------------------------------------------------------
__global__ void fc_kernel(const float* __restrict__ in, const float* __restrict__ w,
                          const float* __restrict__ bias, float* __restrict__ out,
                          int Cin, int Cout){
  int t = blockIdx.x*blockDim.x + threadIdx.x;
  if (t >= BATCH*Cout) return;
  int b = t / Cout, o = t % Cout;
  const float* ip = in + (size_t)b*Cin;
  const float* wp = w  + (size_t)o*Cin;
  float s = bias[o];
  for (int c = 0; c < Cin; ++c) s += ip[c]*wp[c];
  out[t] = s;
}

__global__ void bn_batch_lrelu(float* __restrict__ h, const float* __restrict__ bw,
                               const float* __restrict__ bb, int Cout){
  int o = blockIdx.x*blockDim.x + threadIdx.x;
  if (o >= Cout) return;
  float m = 0.f;
  for (int b = 0; b < BATCH; ++b) m += h[(size_t)b*Cout + o];
  m *= (1.f/BATCH);
  float vr = 0.f;
  for (int b = 0; b < BATCH; ++b){ float d = h[(size_t)b*Cout + o] - m; vr += d*d; }
  vr *= (1.f/BATCH);
  float sc = rsqrtf(vr + 1e-5f) * bw[o];
  float sh = bb[o] - m*sc;
  for (int b = 0; b < BATCH; ++b)
    h[(size_t)b*Cout + o] = lrelu_f(h[(size_t)b*Cout + o]*sc + sh);
}

// ---------------------------------------------------------------------------
template<int CIN, int COUT>
static void edge_layer(const float* xin, const float* w, const float* bw, const float* bb,
                       float* xout, int* idxb, float* xx, float* pd,
                       float* ybuf, float* sums, hipStream_t stream){
  xx_kernel<CIN><<<(BATCH*NPTS + 255)/256, 256, 0, stream>>>(xin, xx);
  pd_kernel<CIN><<<dim3(2, 64, BATCH), 256, 0, stream>>>(xin, xx, pd);
  topk_kernel<<<(BATCH*NPTS + 255)/256, 256, 0, stream>>>(pd, idxb);
  zero_kernel<<<(2048 + 255)/256, 256, 0, stream>>>(sums, 2048);
  edgeconv_gemm<CIN, COUT><<<dim3(NKTOT/128, COUT/64, BATCH), 256, 0, stream>>>(
      xin, idxb, w, ybuf, sums, sums + 1024);
  bn_lrelu_maxk<<<((size_t)BATCH*COUT*NPTS + 255)/256, 256, 0, stream>>>(
      ybuf, sums, sums + 1024, bw, bb, xout, COUT);
}

extern "C" void kernel_launch(void* const* d_in, const int* in_sizes, int n_in,
                              void* d_out, int out_size, void* d_ws, size_t ws_size,
                              hipStream_t stream){
  const float* x   = (const float*)d_in[0];
  const float* c1w = (const float*)d_in[1];
  const float* c2w = (const float*)d_in[2];
  const float* c3w = (const float*)d_in[3];
  const float* c4w = (const float*)d_in[4];
  const float* c5w = (const float*)d_in[5];
  const float* b1w = (const float*)d_in[6];  const float* b1b = (const float*)d_in[7];
  const float* b2w = (const float*)d_in[8];  const float* b2b = (const float*)d_in[9];
  const float* b3w = (const float*)d_in[10]; const float* b3b = (const float*)d_in[11];
  const float* b4w = (const float*)d_in[12]; const float* b4b = (const float*)d_in[13];
  const float* b5w = (const float*)d_in[14]; const float* b5b = (const float*)d_in[15];
  const float* b6w = (const float*)d_in[16]; const float* b6b = (const float*)d_in[17];
  const float* b7w = (const float*)d_in[18]; const float* b7b = (const float*)d_in[19];
  const float* l1w = (const float*)d_in[20]; const float* l1b = (const float*)d_in[21];
  const float* l2w = (const float*)d_in[22]; const float* l2b = (const float*)d_in[23];
  const float* l3w = (const float*)d_in[24]; const float* l3b = (const float*)d_in[25];

  char* ws = (char*)d_ws;
  size_t off = 0;
  auto carve = [&](size_t bytes) -> void* {
    void* p = ws + off; off += (bytes + 255) & ~(size_t)255; return p;
  };
  int*   idxb = (int*)  carve(sizeof(int)  * BATCH*NPTS*KNN);
  float* xx   = (float*)carve(sizeof(float)* BATCH*NPTS);
  float* x1   = (float*)carve(sizeof(float)* BATCH*64 *NPTS);
  float* x2   = (float*)carve(sizeof(float)* BATCH*64 *NPTS);
  float* x3   = (float*)carve(sizeof(float)* BATCH*128*NPTS);
  float* x4   = (float*)carve(sizeof(float)* BATCH*256*NPTS);
  float* sums = (float*)carve(sizeof(float)* 2048);
  float* feat = (float*)carve(sizeof(float)* BATCH*2048);
  float* h1   = (float*)carve(sizeof(float)* BATCH*512);
  float* h2   = (float*)carve(sizeof(float)* BATCH*256);
  float* big  = (float*)carve(sizeof(float)* (size_t)BATCH*256*NKTOT); // pd / y / y5 share
  float* pd   = big;
  float* ybuf = big;
  float* y5   = big;
  (void)in_sizes; (void)n_in; (void)ws_size; (void)out_size;

  edge_layer<3,   64>(x,  c1w, b1w, b1b, x1, idxb, xx, pd, ybuf, sums, stream);
  edge_layer<64,  64>(x1, c2w, b2w, b2b, x2, idxb, xx, pd, ybuf, sums, stream);
  edge_layer<64, 128>(x2, c3w, b3w, b3b, x3, idxb, xx, pd, ybuf, sums, stream);
  edge_layer<128,256>(x3, c4w, b4w, b4b, x4, idxb, xx, pd, ybuf, sums, stream);

  zero_kernel<<<(2048 + 255)/256, 256, 0, stream>>>(sums, 2048);
  conv5_gemm<<<dim3(NPTS/128, 1024/64, BATCH), 256, 0, stream>>>(
      x1, x2, x3, x4, c5w, y5, sums, sums + 1024);
  bn_lrelu_pool<<<(BATCH*1024 + 255)/256, 256, 0, stream>>>(
      y5, sums, sums + 1024, b5w, b5b, feat);

  fc_kernel<<<(BATCH*512 + 255)/256, 256, 0, stream>>>(feat, l1w, l1b, h1, 2048, 512);
  bn_batch_lrelu<<<(512 + 255)/256, 256, 0, stream>>>(h1, b6w, b6b, 512);
  fc_kernel<<<(BATCH*256 + 255)/256, 256, 0, stream>>>(h1, l2w, l2b, h2, 512, 256);
  bn_batch_lrelu<<<(256 + 255)/256, 256, 0, stream>>>(h2, b7w, b7b, 256);
  fc_kernel<<<(BATCH*40 + 255)/256, 256, 0, stream>>>(h2, l3w, l3b, (float*)d_out, 256, 40);
}
